// EdgeLLMAttentionTRTNative_69715909149331
// MI455X (gfx1250) — matrix operand order repaired and checked by reading, not compiled, not run on hardware
//
#include <hip/hip_runtime.h>
#include <hip/hip_bf16.h>

typedef __attribute__((ext_vector_type(16))) __bf16 v16bf;
typedef __attribute__((ext_vector_type(8)))  __bf16 v8bf;
typedef __attribute__((ext_vector_type(8)))  float  v8f;
typedef __attribute__((ext_vector_type(16))) unsigned short v16u;
typedef __attribute__((ext_vector_type(4)))  unsigned int v4ui;
typedef __attribute__((ext_vector_type(8)))  int v8si;
typedef __attribute__((ext_vector_type(4)))  int v4si;
typedef int gv4i __attribute__((vector_size(16)));   // matches builtin param pointee
typedef unsigned short u16;

// ---------------- CDNA5 feature detection ----------------
#if defined(__has_builtin)
#if __has_builtin(__builtin_amdgcn_global_load_async_to_lds_b128) && \
    __has_builtin(__builtin_amdgcn_s_wait_asynccnt)
#define HAS_ASYNC 1
#endif
#if __has_builtin(__builtin_amdgcn_tensor_load_to_lds) && \
    __has_builtin(__builtin_amdgcn_s_wait_tensorcnt)
#define HAS_TDM 1
#endif
#endif
#ifndef HAS_ASYNC
#define HAS_ASYNC 0
#endif
#ifndef HAS_TDM
#define HAS_TDM 0
#endif

// global (AS1) / LDS (AS3) typed pointers for async-to-LDS builtins
#define GVP(p) ((__attribute__((address_space(1))) gv4i*)(p))
#define LVP(p) ((__attribute__((address_space(3))) gv4i*)(p))
#define AS3P(p) ((__attribute__((address_space(3))) void*)(p))

// ---------------- constants ----------------
constexpr int Bc   = 4;
constexpr int Sc   = 512;
constexpr int HIDc = 4096;
constexpr int NHc  = 32;
constexpr int NKVc = 8;
constexpr int Dc   = 128;
constexpr int CAPc = 2048;
constexpr float QK_SCALEc = 0.08838834764831845f; // 1/sqrt(128)
constexpr float EPSc = 1e-6f;

// ---------------- helpers ----------------
__device__ __forceinline__ u16 f2bf(float f) {
  union { float f; unsigned u; } c; c.f = f;
  unsigned u = c.u + 0x7fffu + ((c.u >> 16) & 1u);  // round-to-nearest-even
  return (u16)(u >> 16);
}

__device__ __forceinline__ v16bf mk16(v8bf lo, v8bf hi) {
  v16bf r;
#pragma unroll
  for (int i = 0; i < 8; ++i) { r[i] = lo[i]; r[i + 8] = hi[i]; }
  return r;
}

__device__ __forceinline__ v8f wmma_bf16(v16bf a, v16bf b, v8f c) {
  // D = A(16x32) * B(32x16) + C(16x16 f32)
  return __builtin_amdgcn_wmma_f32_16x16x32_bf16(false, a, false, b, (short)0, c,
                                                 false, false);
}

#if HAS_TDM
// Issue a TDM load of a 32x128 bf16 tile (row stride 128) into LDS.
__device__ __forceinline__ void tdm_tile_32x128(const u16* g, void* lptr) {
  unsigned long long ga = (unsigned long long)(const void*)g;
  unsigned ldsa = (unsigned)(unsigned long long)AS3P(lptr);
  v4ui g0 = { 1u,                                    // count=1, user descriptor
              ldsa,                                  // lds_addr (bytes)
              (unsigned)(ga & 0xffffffffu),          // global_addr[31:0]
              (unsigned)((ga >> 32) & 0x01ffffffu) | (2u << 30) };  // addr[56:32]|type=2
  v8si g1 = { (int)(1u << 16),    // data_size = 2 bytes
              (int)(128u << 16),  // tensor_dim0 = 128 (bits 79:48)
              (int)(2048u << 16), // tensor_dim1 = 2048 (bits 111:80)
              (int)(128u << 16),  // tile_dim0 = 128 (bits 127:112)
              32,                 // tile_dim1 = 32 (bits 143:128)
              128,                // tensor_dim0_stride = 128 (bits 207:160)
              0, 0 };
  v4si gz = { 0, 0, 0, 0 };
#if __clang_major__ >= 23
  v8si gz8 = { 0, 0, 0, 0, 0, 0, 0, 0 };
  __builtin_amdgcn_tensor_load_to_lds(g0, g1, gz, gz, gz8, 0);
#else
  __builtin_amdgcn_tensor_load_to_lds(g0, g1, gz, gz, 0);
#endif
}
#endif

// ---------------- fp32 -> bf16 cast ----------------
__global__ void cast_bf16(const float* __restrict__ in, u16* __restrict__ out, int n) {
  int i = blockIdx.x * blockDim.x + threadIdx.x;
  if (i < n) out[i] = f2bf(in[i]);
}

// ---------------- GEMM: C[M,N](f32) = A[M,K](bf16) * B[K,N](bf16) ----------------
// Block tile 128x128, BK=32, 8 waves each computing 32x64.
// Double-buffered LDS staging via GLOBAL_LOAD_ASYNC_TO_LDS_B128 (ASYNCcnt).
__global__ __launch_bounds__(256) void gemm_bf16(const u16* __restrict__ A,
                                                 const u16* __restrict__ Bm,
                                                 float* __restrict__ C,
                                                 int M, int N, int K) {
  __shared__ __attribute__((aligned(32))) u16 As[2][128 * 48];  // row stride 48 (96B)
  __shared__ __attribute__((aligned(32))) u16 Bs[2][32 * 144];  // row stride 144 (288B)

  const int tid  = threadIdx.x;
  const int lane = tid & 31;
  const int wid  = tid >> 5;
  const int m0 = blockIdx.y * 128, n0 = blockIdx.x * 128;
  const int wm = (wid >> 1) * 32, wn = (wid & 1) * 64;

  v8f acc[2][4];
#pragma unroll
  for (int i = 0; i < 2; ++i)
#pragma unroll
    for (int j = 0; j < 4; ++j) acc[i][j] = (v8f){};

  const int rA = tid >> 1, cA = (tid & 1) * 16;   // A stage: 128 rows x 32 cols
  const int rB = tid >> 3, cB = (tid & 7) * 16;   // B stage: 32 rows x 128 cols
  const int koff = (lane >= 16) ? 8 : 0;

  auto stage = [&](int k0, int buf) {
    const u16* ga = A + (size_t)(m0 + rA) * K + k0 + cA;
    const u16* gb = Bm + (size_t)(k0 + rB) * N + n0 + cB;
#if HAS_ASYNC
    __builtin_amdgcn_global_load_async_to_lds_b128(GVP(ga),
        LVP(&As[buf][rA * 48 + cA]), 0, 0);
    __builtin_amdgcn_global_load_async_to_lds_b128(GVP(ga + 8),
        LVP(&As[buf][rA * 48 + cA + 8]), 0, 0);
    __builtin_amdgcn_global_load_async_to_lds_b128(GVP(gb),
        LVP(&Bs[buf][rB * 144 + cB]), 0, 0);
    __builtin_amdgcn_global_load_async_to_lds_b128(GVP(gb + 8),
        LVP(&Bs[buf][rB * 144 + cB + 8]), 0, 0);
#else
    *(v8bf*)&As[buf][rA * 48 + cA]     = *(const v8bf*)ga;
    *(v8bf*)&As[buf][rA * 48 + cA + 8] = *(const v8bf*)(ga + 8);
    *(v16bf*)&Bs[buf][rB * 144 + cB]   = *(const v16bf*)gb;
    if (k0 + 32 < K) {                 // prefetch next K tile (global_prefetch_b8)
      __builtin_prefetch(ga + 32, 0, 1);
      __builtin_prefetch(gb + (size_t)32 * N, 0, 1);
    }
#endif
  };

  stage(0, 0);
#if HAS_ASYNC
  __builtin_amdgcn_s_wait_asynccnt(0);
#endif
  __syncthreads();

  int buf = 0;
  for (int k0 = 0; k0 < K; k0 += 32) {
    if (k0 + 32 < K) stage(k0 + 32, buf ^ 1);   // overlap next tile with WMMAs

    v16bf af[2];
#pragma unroll
    for (int i = 0; i < 2; ++i) {
      int row = wm + 16 * i + (lane & 15);
      const u16* p = &As[buf][row * 48 + koff];
      af[i] = mk16(*(const v8bf*)p, *(const v8bf*)(p + 16));
    }
    v16bf bfr[4];
#pragma unroll
    for (int j = 0; j < 4; ++j)
#pragma unroll
      for (int s = 0; s < 16; ++s)
        bfr[j][s] = __builtin_bit_cast(__bf16, Bs[buf][(koff + (s & 7) + 16 * (s >> 3)) * 144 + wn + 16 * j + (lane & 15)]);

#pragma unroll
    for (int i = 0; i < 2; ++i)
#pragma unroll
      for (int j = 0; j < 4; ++j) acc[i][j] = wmma_bf16(af[i], bfr[j], acc[i][j]);

#if HAS_ASYNC
    __builtin_amdgcn_s_wait_asynccnt(0);
#endif
    __syncthreads();
    buf ^= 1;
  }

  const int ro = (lane >= 16) ? 8 : 0, co = lane & 15;
#pragma unroll
  for (int i = 0; i < 2; ++i)
#pragma unroll
    for (int j = 0; j < 4; ++j)
#pragma unroll
      for (int v = 0; v < 8; ++v)
        C[(size_t)(m0 + wm + 16 * i + ro + v) * N + n0 + wn + 16 * j + co] =
            acc[i][j][v];
}

// ------ Q prep: RMSNorm + RoPE + scale -> Q^T bf16 [B,NH,D,S] (for S^T = K * Q^T) ------
__global__ void qprep(const float* __restrict__ qproj, const float* __restrict__ rope,
                      const int* __restrict__ kvstart, const float* __restrict__ qnw,
                      u16* __restrict__ qbt) {
  const int bx = blockIdx.x;
  const int h = bx & (NHc - 1);
  const int s = (bx >> 5) & (Sc - 1);
  const int b = bx >> 14;
  const int d = threadIdx.x;
  __shared__ float red[Dc];
  __shared__ float sh[Dc];

  float x = qproj[(size_t)(b * Sc + s) * (NHc * Dc) + h * Dc + d];
  red[d] = x * x;
  __syncthreads();
  for (int o = 64; o > 0; o >>= 1) {
    if (d < o) red[d] += red[d + o];
    __syncthreads();
  }
  float rms = rsqrtf(red[0] * (1.0f / Dc) + EPSc);
  float xn = x * rms * qnw[d];
  sh[d] = xn;
  __syncthreads();

  const int pos = kvstart[b] + s;
  float out;
  if (d < 64) {
    float c = rope[(size_t)pos * Dc + d], sn = rope[(size_t)pos * Dc + 64 + d];
    out = xn * c - sh[d + 64] * sn;
  } else {
    int dd = d - 64;
    float c = rope[(size_t)pos * Dc + dd], sn = rope[(size_t)pos * Dc + 64 + dd];
    out = xn * c + sh[dd] * sn;
  }
  // transposed store: [B,NH,D,S]
  qbt[((size_t)(b * NHc + h) * Dc + d) * Sc + s] = f2bf(out * QK_SCALEc);
}

// ------- KV prep: present_k/v (fp32, d_out) + bf16 mirrors; K gets RMSNorm+RoPE -------
__global__ void kvprep(const float* __restrict__ kproj, const float* __restrict__ vproj,
                       const float* __restrict__ k_cache, const float* __restrict__ v_cache,
                       const float* __restrict__ rope, const int* __restrict__ kvstart,
                       const float* __restrict__ knw,
                       float* __restrict__ pk, float* __restrict__ pv,
                       u16* __restrict__ kb, u16* __restrict__ vb) {
  const int bx = blockIdx.x;
  const int t  = bx & (CAPc - 1);
  const int kv = (bx >> 11) & (NKVc - 1);
  const int b  = bx >> 14;
  const int d  = threadIdx.x;
  const bool isK = (blockIdx.y == 0);
  const int st = kvstart[b];
  const size_t oidx = ((size_t)(b * NKVc + kv) * CAPc + t) * Dc + d;
  __shared__ float red[Dc];
  __shared__ float sh[Dc];

  const int s = t - st;
  if (s >= 0 && s < Sc) {
    if (isK) {
      float x = kproj[(size_t)(b * Sc + s) * (NKVc * Dc) + kv * Dc + d];
      red[d] = x * x;
      __syncthreads();
      for (int o = 64; o > 0; o >>= 1) {
        if (d < o) red[d] += red[d + o];
        __syncthreads();
      }
      float rms = rsqrtf(red[0] * (1.0f / Dc) + EPSc);
      float xn = x * rms * knw[d];
      sh[d] = xn;
      __syncthreads();
      const int pos = st + s;
      float out;
      if (d < 64) {
        float c = rope[(size_t)pos * Dc + d], sn = rope[(size_t)pos * Dc + 64 + d];
        out = xn * c - sh[d + 64] * sn;
      } else {
        int dd = d - 64;
        float c = rope[(size_t)pos * Dc + dd], sn = rope[(size_t)pos * Dc + 64 + dd];
        out = xn * c + sh[dd] * sn;
      }
      pk[oidx] = out; kb[oidx] = f2bf(out);
    } else {
      float xv = vproj[(size_t)(b * Sc + s) * (NKVc * Dc) + kv * Dc + d];
      pv[oidx] = xv; vb[oidx] = f2bf(xv);
    }
  } else {
    if (isK) { float x = k_cache[oidx]; pk[oidx] = x; kb[oidx] = f2bf(x); }
    else     { float x = v_cache[oidx]; pv[oidx] = x; vb[oidx] = f2bf(x); }
  }
}

// ---------------- flash attention: per (b,h,128 q-rows) workgroup ----------------
// S^T = K * Q^T so both K and V tiles are row-major 32x128 (TDM-loadable, no transpose).
// The S^T accumulator layout IS the A-fragment layout for P, so softmax is register-only
// (one __shfl_xor(16) pair reduction per tile; row broadcasts via __shfl/ds_bpermute).
__global__ __launch_bounds__(256) void attn_kernel(const u16* __restrict__ qbt,
                                                   const u16* __restrict__ kb,
                                                   const u16* __restrict__ vb,
                                                   const int* __restrict__ kvstart,
                                                   u16* __restrict__ ctx) {
  __shared__ __attribute__((aligned(32))) u16 Kt[2][32 * 128];  // K tile [t][d]
  __shared__ __attribute__((aligned(32))) u16 Vt[2][32 * 128];  // V tile [t][d]

  const int tid = threadIdx.x, lane = tid & 31, wid = tid >> 5;
  const int qbase = blockIdx.x * 128;
  const int h = blockIdx.y, b = blockIdx.z;
  const int kvh = h >> 2;  // NH/NKV = 4
  const int st = kvstart[b];
  const int mcol = lane & 15;          // q-row within wave tile (column of S^T)
  const int ho = (lane >> 4) * 8;      // 0 or 8: k-chunk / accum-row offset

  // Q^T B-fragments (held in registers entire kernel): B[d][m]
  const u16* qtp = qbt + (size_t)(b * NHc + h) * Dc * Sc;  // [D][S]
  v16bf qfrag[4];
#pragma unroll
  for (int kk = 0; kk < 4; ++kk)
#pragma unroll
    for (int s = 0; s < 16; ++s)
      qfrag[kk][s] = __builtin_bit_cast(__bf16, qtp[(size_t)(32 * kk + ho + (s & 7) + 16 * (s >> 3)) * Sc + qbase + wid * 16 + mcol]);

  v8f acc[8];
#pragma unroll
  for (int j = 0; j < 8; ++j) acc[j] = (v8f){};
  float m_run = -3.0e38f, l_run = 0.0f;

  const u16* kbp = kb + (size_t)(b * NKVc + kvh) * CAPc * Dc;
  const u16* vbp = vb + (size_t)(b * NKVc + kvh) * CAPc * Dc;

  const int ttr  = tid >> 3;        // 0..31
  const int dblk = (tid & 7) * 16;  // 0..112
  const int maxpos = st + qbase + 127;

  auto stage = [&](int t0, int buf) {
#if HAS_TDM
    if (wid == 0) {
      tdm_tile_32x128(kbp + (size_t)t0 * Dc, &Kt[buf][0]);
      tdm_tile_32x128(vbp + (size_t)t0 * Dc, &Vt[buf][0]);
    }
#elif HAS_ASYNC
    const u16* ksrc = kbp + (size_t)(t0 + ttr) * Dc + dblk;
    const u16* vsrc = vbp + (size_t)(t0 + ttr) * Dc + dblk;
    __builtin_amdgcn_global_load_async_to_lds_b128(GVP(ksrc),
        LVP(&Kt[buf][ttr * 128 + dblk]), 0, 0);
    __builtin_amdgcn_global_load_async_to_lds_b128(GVP(ksrc + 8),
        LVP(&Kt[buf][ttr * 128 + dblk + 8]), 0, 0);
    __builtin_amdgcn_global_load_async_to_lds_b128(GVP(vsrc),
        LVP(&Vt[buf][ttr * 128 + dblk]), 0, 0);
    __builtin_amdgcn_global_load_async_to_lds_b128(GVP(vsrc + 8),
        LVP(&Vt[buf][ttr * 128 + dblk + 8]), 0, 0);
#else
    *(v16u*)&Kt[buf][ttr * 128 + dblk] = *(const v16u*)(kbp + (size_t)(t0 + ttr) * Dc + dblk);
    *(v16u*)&Vt[buf][ttr * 128 + dblk] = *(const v16u*)(vbp + (size_t)(t0 + ttr) * Dc + dblk);
#endif
  };

  stage(0, 0);
#if HAS_TDM
  if (wid == 0) __builtin_amdgcn_s_wait_tensorcnt(0);
#elif HAS_ASYNC
  __builtin_amdgcn_s_wait_asynccnt(0);
#endif
  __syncthreads();

  int buf = 0;
  for (int t0 = 0; t0 <= maxpos; t0 += 32) {
    if (t0 + 32 <= maxpos) stage(t0 + 32, buf ^ 1);  // overlap next K/V tiles

    // ---- S^T tiles: sA = t0..t0+15, sB = t0+16..t0+31 (rows t, cols m) ----
    v8f sA = (v8f){}, sB = (v8f){};
#pragma unroll
    for (int kk = 0; kk < 4; ++kk) {
      const u16* pa = &Kt[buf][mcol * 128 + 32 * kk + ho];
      v16bf aA = mk16(*(const v8bf*)pa, *(const v8bf*)(pa + 16));
      sA = wmma_bf16(aA, qfrag[kk], sA);
      const u16* pb = &Kt[buf][(16 + mcol) * 128 + 32 * kk + ho];
      v16bf aB = mk16(*(const v8bf*)pb, *(const v8bf*)(pb + 16));
      sB = wmma_bf16(aB, qfrag[kk], sB);
    }

    // ---- register-resident online softmax (lane owns q-row mcol; t split with lane^16) ----
    const int pos = st + qbase + wid * 16 + mcol;
    const int tA = t0 + ho, tB = t0 + 16 + ho;
    float pA[8], pB[8];
    float tmax = -1.0e30f;
#pragma unroll
    for (int v = 0; v < 8; ++v) {
      pA[v] = (tA + v <= pos) ? sA[v] : -1.0e30f;
      pB[v] = (tB + v <= pos) ? sB[v] : -1.0e30f;
      tmax = fmaxf(tmax, fmaxf(pA[v], pB[v]));
    }
    tmax = fmaxf(tmax, __shfl_xor(tmax, 16));
    float mnew = fmaxf(m_run, tmax);
    float f = __expf(m_run - mnew);
    float rsum = 0.0f;
#pragma unroll
    for (int v = 0; v < 8; ++v) {
      pA[v] = (tA + v <= pos) ? __expf(pA[v] - mnew) : 0.0f;
      pB[v] = (tB + v <= pos) ? __expf(pB[v] - mnew) : 0.0f;
      rsum += pA[v] + pB[v];
    }
    rsum += __shfl_xor(rsum, 16);
    l_run = l_run * f + rsum;
    m_run = mnew;

    // ---- P A-fragment directly from registers (S^T accum layout == A layout) ----
    v16bf pfrag;
#pragma unroll
    for (int v = 0; v < 8; ++v) {
      pfrag[v]     = __builtin_bit_cast(__bf16, f2bf(pA[v]));
      pfrag[v + 8] = __builtin_bit_cast(__bf16, f2bf(pB[v]));
    }

    // ---- rescale out accumulator (row factors via ds_bpermute) and add P*V ----
    float f8[8];
#pragma unroll
    for (int v = 0; v < 8; ++v) f8[v] = __shfl(f, v + ho);
#pragma unroll
    for (int j = 0; j < 8; ++j) {
#pragma unroll
      for (int v = 0; v < 8; ++v) acc[j][v] *= f8[v];
      v16bf bv;
#pragma unroll
      for (int s = 0; s < 16; ++s)
        bv[s] = __builtin_bit_cast(__bf16, Vt[buf][(ho + (s & 7) + 16 * (s >> 3)) * 128 + 16 * j + mcol]);
      acc[j] = wmma_bf16(pfrag, bv, acc[j]);
    }

#if HAS_TDM
    if (wid == 0) __builtin_amdgcn_s_wait_tensorcnt(0);
#elif HAS_ASYNC
    __builtin_amdgcn_s_wait_asynccnt(0);
#endif
    __syncthreads();
    buf ^= 1;
  }

  // ---- epilogue: 1/l per row via shuffle broadcast, store bf16 context ----
  float linv[8];
#pragma unroll
  for (int v = 0; v < 8; ++v) linv[v] = 1.0f / __shfl(l_run, v + ho);
#pragma unroll
  for (int j = 0; j < 8; ++j)
#pragma unroll
    for (int v = 0; v < 8; ++v) {
      int srow = qbase + wid * 16 + ho + v;
      ctx[(size_t)(b * Sc + srow) * HIDc + h * Dc + j * 16 + mcol] =
          f2bf(acc[j][v] * linv[v]);
    }
}

// ---------------- host-side launch ----------------
extern "C" void kernel_launch(void* const* d_in, const int* in_sizes, int n_in,
                              void* d_out, int out_size, void* d_ws, size_t ws_size,
                              hipStream_t stream) {
  (void)in_sizes; (void)n_in; (void)out_size; (void)ws_size;

  const float* hidden  = (const float*)d_in[0];
  const float* k_cache = (const float*)d_in[1];
  const float* v_cache = (const float*)d_in[2];
  const float* rope    = (const float*)d_in[3];
  const int*   kvstart = (const int*)d_in[5];
  const float* Wq      = (const float*)d_in[6];
  const float* Wk      = (const float*)d_in[7];
  const float* Wv      = (const float*)d_in[8];
  const float* Wo      = (const float*)d_in[9];
  const float* qnw     = (const float*)d_in[10];
  const float* knw     = (const float*)d_in[11];

  char* ws = (char*)d_ws;
  u16*   hsb = (u16*)(ws + 0);            // 2048*4096 bf16
  u16*   wqb = (u16*)(ws + 16777216);     // 4096*4096
  u16*   wkb = (u16*)(ws + 50331648);     // 4096*1024
  u16*   wvb = (u16*)(ws + 58720256);     // 4096*1024
  u16*   wob = (u16*)(ws + 67108864);     // 4096*4096
  float* qp  = (float*)(ws + 100663296);  // 2048*4096 f32
  float* kp  = (float*)(ws + 134217728);  // 2048*1024 f32
  float* vp  = (float*)(ws + 142606336);  // 2048*1024 f32
  u16*   qbb = (u16*)(ws + 150994944);    // Q^T [B,NH,D,S]
  u16*   kbb = (u16*)(ws + 167772160);    // [B,NKV,CAP,D]
  u16*   vbb = (u16*)(ws + 184549376);    // [B,NKV,CAP,D]
  u16*   ctx = (u16*)(ws + 201326592);    // [B,S,NH*D]

  float* out = (float*)d_out;
  float* pk  = out + (size_t)Bc * Sc * HIDc;          // 8,388,608
  float* pv  = pk + (size_t)Bc * NKVc * CAPc * Dc;    // +8,388,608

  // fp32 -> bf16 casts
  { int n = Bc * Sc * HIDc;   cast_bf16<<<(n + 255) / 256, 256, 0, stream>>>(hidden, hsb, n); }
  { int n = HIDc * NHc * Dc;  cast_bf16<<<(n + 255) / 256, 256, 0, stream>>>(Wq, wqb, n); }
  { int n = HIDc * NKVc * Dc; cast_bf16<<<(n + 255) / 256, 256, 0, stream>>>(Wk, wkb, n); }
  { int n = HIDc * NKVc * Dc; cast_bf16<<<(n + 255) / 256, 256, 0, stream>>>(Wv, wvb, n); }
  { int n = NHc * Dc * HIDc;  cast_bf16<<<(n + 255) / 256, 256, 0, stream>>>(Wo, wob, n); }

  // QKV projections
  gemm_bf16<<<dim3(32, 16), 256, 0, stream>>>(hsb, wqb, qp, 2048, 4096, 4096);
  gemm_bf16<<<dim3(8, 16), 256, 0, stream>>>(hsb, wkb, kp, 2048, 1024, 4096);
  gemm_bf16<<<dim3(8, 16), 256, 0, stream>>>(hsb, wvb, vp, 2048, 1024, 4096);

  // norm + rope + cache update
  qprep<<<dim3(Bc * Sc * NHc), 128, 0, stream>>>(qp, rope, kvstart, qnw, qbb);
  kvprep<<<dim3(Bc * NKVc * CAPc, 2), 128, 0, stream>>>(kp, vp, k_cache, v_cache, rope,
                                                        kvstart, knw, pk, pv, kbb, vbb);

  // flash attention
  attn_kernel<<<dim3(Sc / 128, NHc, Bc), 256, 0, stream>>>(qbb, kbb, vbb, kvstart, ctx);

  // O projection
  gemm_bf16<<<dim3(32, 16), 256, 0, stream>>>(ctx, wob, out, 2048, 4096, 4096);
}
